// ConditionalRBM_385
// MI455X (gfx1250) — matrix operand order repaired and checked by reading, not compiled, not run on hardware
//
#include <hip/hip_runtime.h>
#include <hip/hip_bf16.h>

// ---------------------------------------------------------------------------
// Conditional RBM Gibbs sampler for MI455X (gfx1250, wave32, WMMA).
//
// Strategy: block = 256 threads (8 waves) owns BT=32 batch rows and runs all
// `steps` Gibbs iterations with W resident in LDS (two bf16 layouts so both
// v@W and h@W^T use clean ds_load_b128 fragment fetches). Each wave computes
// a 16x64 slab of the [32,256] activation tile with v_wmma_f32_16x16x32_bf16.
// Row reductions (asum/va/vb) use shfl_xor butterflies + LDS float atomics.
// Sigmoids use v_exp + v_rcp (no IEEE divide expansion) — sampler precision
// does not need correctly-rounded division.
// ---------------------------------------------------------------------------

typedef __attribute__((ext_vector_type(16))) __bf16 v16bf;
typedef __attribute__((ext_vector_type(8)))  float  v8f;

#define BQ      8192
#define BT      32          // batch rows per block
#define WSTR    264         // padded LDS row stride in elements (528B = 33*16B)
#define TEND    1.0f
#define TSTART  5.0f
#define FALLOFF 0.1f

__device__ __forceinline__ float rng01(unsigned a, unsigned b, unsigned c) {
  unsigned x = a * 0x9E3779B9u ^ (b + 0x85EBCA6Bu) * 0xC2B2AE35u ^ (c * 0x27D4EB2Fu);
  x ^= x >> 16; x *= 0x7FEB352Du;
  x ^= x >> 15; x *= 0x846CA68Bu;
  x ^= x >> 16;
  return (float)(x >> 8) * (1.0f / 16777216.0f);
}
// Fast sigmoid: v_exp_f32 + v_add + v_rcp_f32 (avoids div_scale/div_fmas chain).
__device__ __forceinline__ float sigm(float x) {
  return __builtin_amdgcn_rcpf(1.0f + __expf(-x));
}
__device__ __forceinline__ unsigned short f2bf(float f) {
  unsigned u = __float_as_uint(f);
  return (unsigned short)((u + 0x7FFFu + ((u >> 16) & 1u)) >> 16);
}
__device__ __forceinline__ float bf2f(unsigned short h) {
  return __uint_as_float(((unsigned)h) << 16);
}

// Load a bf16 WMMA fragment (16 elems/lane) from LDS as two 16B chunks.
// off2 = element offset of second uint4 chunk relative to p.
__device__ __forceinline__ v16bf ld_frag(const unsigned short* p, int off2) {
  union { v16bf v; uint4 q[2]; } u;
  u.q[0] = *(const uint4*)(p);
  u.q[1] = *(const uint4*)(p + off2);
  return u.v;
}

// acc[4] += A(tile rows mtile*16..+15, K=0..255) x B(n = nt0*16 .. nt0*16+63)
// A layout (16-bit 16x32): lane m = lane&15; half=lane>>4 selects K sub-ranges
//   {half*8..half*8+7, 16+half*8..16+half*8+7}
// B layout (16-bit 32x16): lane n = lane&15; half selects K sub-ranges
//   {half*8..half*8+7, 16+half*8..16+half*8+7}
__device__ __forceinline__ void gemm_tile(const unsigned short* Atile,
                                          const unsigned short* Bmat,
                                          int mtile, int nt0, int lane,
                                          v8f acc[4]) {
  const int ln = lane & 15;
  const int hf = lane >> 4;
  const int arow = mtile * 16 + ln;
#pragma unroll
  for (int kb = 0; kb < 8; ++kb) {
    const int k0 = kb * 32;
    v16bf a = ld_frag(Atile + arow * WSTR + k0 + hf * 8, 16);
#pragma unroll
    for (int j = 0; j < 4; ++j) {
      const int n = (nt0 + j) * 16 + ln;
      v16bf b = ld_frag(Bmat + n * WSTR + k0 + hf * 8, 16);
      acc[j] = __builtin_amdgcn_wmma_f32_16x16x32_bf16(
          false, a, false, b, (short)0, acc[j], false, false);
    }
  }
}

// ---------------------------------------------------------------------------
// Kernel 1: conditioner MLP -> b_mod [B,256], c_mod [B,256] in workspace.
// One block per batch row.
// ---------------------------------------------------------------------------
__global__ __launch_bounds__(256)
void cond_kernel(const float* __restrict__ cond, const float* __restrict__ bvec,
                 const float* __restrict__ cvec, const float* __restrict__ fc1w,
                 const float* __restrict__ fc1b, const float* __restrict__ fc2w,
                 const float* __restrict__ fc2b, float* __restrict__ bmod,
                 float* __restrict__ cmod) {
  __shared__ float hid[64];
  const int row = blockIdx.x;
  const int tid = threadIdx.x;
  if (tid < 64) {
    float s = fc1b[tid];
    const float* wr = fc1w + tid * 16;
    const float* xr = cond + (size_t)row * 16;
#pragma unroll
    for (int k = 0; k < 16; ++k) s += wr[k] * xr[k];
    hid[tid] = tanhf(s);
  }
  __syncthreads();
  const int o = tid;  // 0..255
  float g  = fc2b[o];
  float be = fc2b[256 + o];
  float g2 = fc2b[512 + o];
  float be2 = fc2b[768 + o];
  const float* w0 = fc2w + (size_t)o * 64;
  const float* w1 = fc2w + (size_t)(256 + o) * 64;
  const float* w2 = fc2w + (size_t)(512 + o) * 64;
  const float* w3 = fc2w + (size_t)(768 + o) * 64;
#pragma unroll 4
  for (int k = 0; k < 64; ++k) {
    float h = hid[k];
    g  += w0[k] * h;
    be += w1[k] * h;
    g2 += w2[k] * h;
    be2 += w3[k] * h;
  }
  bmod[(size_t)row * 256 + o] = (1.0f + g) * bvec[o] + be;
  cmod[(size_t)row * 256 + o] = (1.0f + g2) * cvec[o] + be2;
}

// ---------------------------------------------------------------------------
// Kernel 2: the 32-step Gibbs chain. W resident in LDS (both orientations).
// ---------------------------------------------------------------------------
__global__ __launch_bounds__(256)
void gibbs_kernel(const float* __restrict__ W, const float* __restrict__ bmod,
                  const float* __restrict__ cmod, const int* __restrict__ stepsp,
                  float* __restrict__ out) {
  extern __shared__ char smem[];
  unsigned short* WT = (unsigned short*)smem;      // [256][WSTR]: WT[n*WSTR+k] = W[k][n]  (B for v@W)
  unsigned short* WR = WT + 256 * WSTR;            // [256][WSTR]: WR[n*WSTR+k] = W[n][k]  (B for h@W^T)
  unsigned short* tV = WR + 256 * WSTR;            // [BT][WSTR] bf16, holds v2 (== v_eff of next step)
  unsigned short* tH = tV + BT * WSTR;             // [BT][WSTR] bf16, holds h
  float* s0cur = (float*)(tH + BT * WSTR);         // [BT]
  float* bsum  = s0cur + BT;                       // [BT]
  float* redA  = bsum + BT;                        // [BT] sum a
  float* redVA = redA + BT;                        // [BT] sum v*a
  float* redVB = redVA + BT;                       // [BT] sum v*b_mod

  const int tid  = threadIdx.x;
  const int lane = tid & 31;
  const int wid  = tid >> 5;
  const int mtile = wid >> 2;          // 0..1  (16 rows each)
  const int nt0   = (wid & 3) * 4;     // 4 N-tiles of 16 per wave
  const int ln = lane & 15;
  const int hf = lane >> 4;
  const int grow0 = blockIdx.x * BT;
  const int steps = *stepsp;

  // Stage W into LDS in both orientations (bf16).
  for (int i = tid; i < 256 * 256; i += 256) {
    const int r = i >> 8, c = i & 255;
    const unsigned short w = f2bf(W[i]);
    WR[r * WSTR + c] = w;
    WT[c * WSTR + r] = w;
  }
  // Init v ~ Bernoulli(0.5); s0 = 1 means v_eff(step0) = v = tV contents.
  for (int i = tid; i < BT * 256; i += 256) {
    const int r = i >> 8, c = i & 255;
    const float u = rng01(0xA5A5A5A5u, (unsigned)(grow0 + r), (unsigned)c);
    tV[r * WSTR + c] = (u < 0.5f) ? 0x3F80 : 0x0000;
  }
  if (tid < BT) {
    s0cur[tid] = 1.0f;
    float s = 0.0f;
    const float* bp = bmod + (size_t)(grow0 + tid) * 256;
    for (int c = 0; c < 256; ++c) s += bp[c];
    bsum[tid] = s;
  }
  __syncthreads();

  for (int t = 0; t < steps; ++t) {
    const float sf = sigm(-FALLOFF * ((float)t - (float)steps * 0.5f));
    const float T = TEND + (TSTART - TEND) * sf;
    const float invT = __builtin_amdgcn_rcpf(T);

    if (tid < BT) { redA[tid] = 0.0f; redVA[tid] = 0.0f; redVB[tid] = 0.0f; }

    // ---- GEMM1: acc = v_eff @ W ----
    v8f acc[4];
#pragma unroll
    for (int j = 0; j < 4; ++j)
#pragma unroll
      for (int e = 0; e < 8; ++e) acc[j][e] = 0.0f;
    gemm_tile(tV, WT, mtile, nt0, lane, acc);

    // Sample h -> tH. C layout: VGPR r, lane: row = mtile*16 + hf*8 + r, n = ntile*16 + ln.
#pragma unroll
    for (int j = 0; j < 4; ++j) {
      const int n = (nt0 + j) * 16 + ln;
#pragma unroll
      for (int r = 0; r < 8; ++r) {
        const int row = mtile * 16 + hf * 8 + r;
        const int grow = grow0 + row;
        const float x = acc[j][r] + cmod[(size_t)grow * 256 + n];
        const float p = sigm(x * invT);
        const float u = rng01((unsigned)(t * 4 + 0), (unsigned)grow, (unsigned)n);
        tH[row * WSTR + n] = (u < p) ? 0x3F80 : 0x0000;
      }
    }
    __syncthreads();  // tH visible; reduction zeros done

    // ---- GEMM2: a = h @ W^T ----
#pragma unroll
    for (int j = 0; j < 4; ++j)
#pragma unroll
      for (int e = 0; e < 8; ++e) acc[j][e] = 0.0f;
    gemm_tile(tH, WR, mtile, nt0, lane, acc);

    // Per-lane: row partial sums (asum, v*a, v*b_mod), sample v2 -> tV.
    // v (true) = s0 ? v2_old : 1 - v2_old (flip involution).
    float pA[8], pVA[8], pVB[8];
#pragma unroll
    for (int r = 0; r < 8; ++r) { pA[r] = 0.0f; pVA[r] = 0.0f; pVB[r] = 0.0f; }
#pragma unroll
    for (int j = 0; j < 4; ++j) {
      const int n = (nt0 + j) * 16 + ln;
#pragma unroll
      for (int r = 0; r < 8; ++r) {
        const int row = mtile * 16 + hf * 8 + r;
        const int grow = grow0 + row;
        const float a  = acc[j][r];
        const float bm = bmod[(size_t)grow * 256 + n];
        const float v2 = bf2f(tV[row * WSTR + n]);
        const float s0 = s0cur[row];
        const float vv = s0 * v2 + (1.0f - s0) * (1.0f - v2);
        pA[r]  += a;
        pVA[r] += vv * a;
        pVB[r] += vv * bm;
        const float pv = sigm((a + bm) * invT);
        const float u = rng01((unsigned)(t * 4 + 2), (unsigned)grow, (unsigned)n);
        tV[row * WSTR + n] = (u < pv) ? 0x3F80 : 0x0000;  // raw v2 == v_eff of next step
      }
    }
    // Butterfly-reduce across the 16 lanes that share this row set.
#pragma unroll
    for (int m = 1; m < 16; m <<= 1) {
#pragma unroll
      for (int r = 0; r < 8; ++r) {
        pA[r]  += __shfl_xor(pA[r],  m, 32);
        pVA[r] += __shfl_xor(pVA[r], m, 32);
        pVB[r] += __shfl_xor(pVB[r], m, 32);
      }
    }
    if (ln == 0) {
#pragma unroll
      for (int r = 0; r < 8; ++r) {
        const int row = mtile * 16 + hf * 8 + r;
        atomicAdd(&redA[row],  pA[r]);
        atomicAdd(&redVA[row], pVA[r]);
        atomicAdd(&redVB[row], pVB[r]);
      }
    }
    __syncthreads();

    // dE per row; sample new s0 (used by NEXT step's flip and final output).
    if (tid < BT) {
      const float dE = -bsum[tid] - redA[tid] + 2.0f * redVB[tid] + 2.0f * redVA[tid];
      const float p = sigm(dE * invT);
      const float u = rng01((unsigned)(t * 4 + 1), (unsigned)(grow0 + tid), 0u);
      s0cur[tid] = (u < p) ? 1.0f : 0.0f;
    }
    __syncthreads();
  }

  // Output v = flip(v2, s0_final), f32.
  for (int i = tid; i < BT * 256; i += 256) {
    const int r = i >> 8, c = i & 255;
    const float v2 = bf2f(tV[r * WSTR + c]);
    const float s0 = s0cur[r];
    out[(size_t)(grow0 + r) * 256 + c] = s0 * v2 + (1.0f - s0) * (1.0f - v2);
  }
  // temps tail (block 0 only)
  if (blockIdx.x == 0 && tid < steps) {
    const float sf = 1.0f / (1.0f + __expf(FALLOFF * ((float)tid - (float)steps * 0.5f)));
    out[(size_t)BQ * 256 + tid] = TEND + (TSTART - TEND) * sf;
  }
}

extern "C" void kernel_launch(void* const* d_in, const int* in_sizes, int n_in,
                              void* d_out, int out_size, void* d_ws, size_t ws_size,
                              hipStream_t stream) {
  (void)in_sizes; (void)n_in; (void)out_size; (void)ws_size;
  const float* cond = (const float*)d_in[0];
  const float* W    = (const float*)d_in[1];
  const float* b    = (const float*)d_in[2];
  const float* c    = (const float*)d_in[3];
  const float* fc1w = (const float*)d_in[4];
  const float* fc1b = (const float*)d_in[5];
  const float* fc2w = (const float*)d_in[6];
  const float* fc2b = (const float*)d_in[7];
  const int* steps  = (const int*)d_in[8];
  float* out = (float*)d_out;

  float* bmod = (float*)d_ws;                    // [8192,256] f32
  float* cmod = bmod + (size_t)BQ * 256;         // [8192,256] f32

  cond_kernel<<<BQ, 256, 0, stream>>>(cond, b, c, fc1w, fc1b, fc2w, fc2b, bmod, cmod);

  const size_t smem = (size_t)(2 * 256 * WSTR + 2 * BT * WSTR) * sizeof(unsigned short)
                    + (size_t)(5 * BT) * sizeof(float);
  gibbs_kernel<<<BQ / BT, 256, smem, stream>>>(W, bmod, cmod, steps, out);
}
